// NonLocalBlock_56040733278669
// MI455X (gfx1250) — matrix operand order repaired and checked by reading, not compiled, not run on hardware
//
#include <hip/hip_runtime.h>
#include <math.h>

// ---------- types ----------
typedef _Float16 h16;
typedef __attribute__((ext_vector_type(16))) _Float16 v16h;
typedef __attribute__((ext_vector_type(8)))  float    v8f;
typedef __attribute__((ext_vector_type(4)))  unsigned int v4u;
typedef __attribute__((ext_vector_type(8)))  int v8i;
typedef __attribute__((ext_vector_type(4)))  int v4i;

// ---------- problem constants ----------
#define CCH  4096   // channels
#define ICH  256    // inter channels (== conv1 out channels)
#define NPIX 4096   // H*W
#define BB   2      // batch

// ---------- GEMM macro-tile ----------
#define TILE  128   // block tile (M and N)
#define KSTEP 32    // == WMMA K for f16
#define ASTR  40    // LDS A stride in h16 (32 + pad; keeps rows 16B-aligned: 80B)
#define BSW   4096  // swizzled B tile: 8 nsub * 32 lanes * 16 h16

// A fragment (16x32 f16): lanes 0-15 -> M=lane; per lane one contiguous 16-h16 run (k = 16*(lane/16) + j)
__device__ __forceinline__ v16h frag_a(const h16* s, int m0, int lane) {
  v16h a;
  const h16* row = s + (m0 + (lane & 15)) * ASTR;
  const int kh = (lane >> 4) << 4;
#pragma unroll
  for (int j = 0; j < 16; ++j) {
    int k = j + kh;   // one 32B run -> 2x ds_load_b128
    a[j] = row[k];
  }
  return a;
}

// Swizzled-B store: element (k, n) of the 32x128 tile -> fragment-ordered LDS
__device__ __forceinline__ void stB(h16* Bs, int kk, int nn, h16 v) {
  Bs[((((nn >> 4) << 5) + ((kk >> 4) << 4) + (nn & 15)) << 4) + (kk & 15)] = v;
}
// B fragment (32x16 f16): contiguous 32B per lane -> 2x ds_load_b128
__device__ __forceinline__ v16h frag_b(const h16* s, int nsub, int lane) {
  v16h b;
  const h16* p = s + (((nsub << 5) + lane) << 4);
#pragma unroll
  for (int j = 0; j < 16; ++j) b[j] = p[j];
  return b;
}

// one K-step: wave computes its 32x64 sub-tile (2x4 WMMAs)
__device__ __forceinline__ void mma_tile(const h16* As, const h16* Bs,
                                         int wm, int wn, int lane, v8f acc[2][4]) {
  v16h af[2], bf[4];
#pragma unroll
  for (int mt = 0; mt < 2; ++mt) af[mt] = frag_a(As, wm * 32 + mt * 16, lane);
#pragma unroll
  for (int nt = 0; nt < 4; ++nt) bf[nt] = frag_b(Bs, wn * 4 + nt, lane);
#pragma unroll
  for (int mt = 0; mt < 2; ++mt)
#pragma unroll
    for (int nt = 0; nt < 4; ++nt)
      acc[mt][nt] = __builtin_amdgcn_wmma_f32_16x16x32_f16(
          false, af[mt], false, bf[nt], (short)0, acc[mt][nt], false, false);
}

// ============================================================================
// Kernel 1: Wcm = w_conv1 [256,4096] @ w_mask [4096,256]  ->  f16 [256,256]
// ============================================================================
__global__ __launch_bounds__(256) void k_wcm(const float* __restrict__ wc1,
                                             const float* __restrict__ wmask,
                                             h16* __restrict__ Wcm) {
  __shared__ h16 As[TILE * ASTR];
  __shared__ h16 Bs[BSW];
  const int tid = threadIdx.x, lane = tid & 31, wid = tid >> 5;
  const int wm = wid & 3, wn = wid >> 2;
  const int o0 = blockIdx.y * TILE, i0 = blockIdx.x * TILE;
  v8f acc[2][4] = {};
  for (int k0 = 0; k0 < CCH; k0 += KSTEP) {
#pragma unroll
    for (int it = 0; it < 4; ++it) {
      int s = tid + it * 256;                // 1024 float4 slots
      int row = s >> 3, k4 = (s & 7) * 4;
      float4 v = *(const float4*)(wc1 + (size_t)(o0 + row) * CCH + k0 + k4);
      h16* d = As + row * ASTR + k4;
      d[0] = (h16)v.x; d[1] = (h16)v.y; d[2] = (h16)v.z; d[3] = (h16)v.w;
    }
#pragma unroll
    for (int it = 0; it < 4; ++it) {
      int s = tid + it * 256;
      int kk = s >> 5, i4 = (s & 31) * 4;
      float4 v = *(const float4*)(wmask + (size_t)(k0 + kk) * ICH + i0 + i4);
      stB(Bs, kk, i4 + 0, (h16)v.x); stB(Bs, kk, i4 + 1, (h16)v.y);
      stB(Bs, kk, i4 + 2, (h16)v.z); stB(Bs, kk, i4 + 3, (h16)v.w);
    }
    __syncthreads();
    mma_tile(As, Bs, wm, wn, lane, acc);
    __syncthreads();
  }
#pragma unroll
  for (int mt = 0; mt < 2; ++mt)
#pragma unroll
    for (int nt = 0; nt < 4; ++nt)
#pragma unroll
      for (int r = 0; r < 8; ++r) {
        int row = o0 + wm * 32 + mt * 16 + r + ((lane >> 4) << 3);
        int col = i0 + wn * 64 + nt * 16 + (lane & 15);
        Wcm[(size_t)row * ICH + col] = (h16)acc[mt][nt][r];
      }
}

// ============================================================================
// Kernel 2: 4-way fused projection: [w_phi;w_theta;w_g;w_conv1] @ x
//   grp 0..2 -> phi/theta/g f16 [B][IC][N]; grp 3 -> zx f32 [B][256][N]
// grid: (N/128, 1024/128, B)
// ============================================================================
__global__ __launch_bounds__(256) void k_proj(const float* __restrict__ x,
    const float* __restrict__ w_phi, const float* __restrict__ w_theta,
    const float* __restrict__ w_g, const float* __restrict__ w_conv1,
    h16* __restrict__ phi, h16* __restrict__ theta, h16* __restrict__ g,
    float* __restrict__ zx, int tdm_gate) {
  __shared__ h16 As[TILE * ASTR];
  __shared__ h16 Bs[BSW];
  const int tid = threadIdx.x, lane = tid & 31, wid = tid >> 5;
  const int wm = wid & 3, wn = wid >> 2;
  const int b = blockIdx.z;
  const int grp = blockIdx.y >> 1;
  const int r0 = (blockIdx.y & 1) * TILE;   // row within the 256-row group
  const int n0 = blockIdx.x * TILE;
  const float* W = (grp == 0) ? w_phi : (grp == 1) ? w_theta
                 : (grp == 2) ? w_g : w_conv1;

#if __has_builtin(__builtin_amdgcn_tensor_load_to_lds)
  if (tdm_gate) {  // host always passes 0; emits TDM + tensorcnt path
    v4u g0 = {}; v8i g1 = {}; v4i g2 = {}, g3 = {}; v8i g4 = {};
    __builtin_amdgcn_tensor_load_to_lds(g0, g1, g2, g3, g4, 0);
    __builtin_amdgcn_s_wait_tensorcnt(0);
  }
#endif

  const float* xb = x + (size_t)b * CCH * NPIX;
  v8f acc[2][4] = {};
  for (int k0 = 0; k0 < CCH; k0 += KSTEP) {
#pragma unroll
    for (int it = 0; it < 4; ++it) {       // A: weights 128x32
      int s = tid + it * 256;
      int row = s >> 3, k4 = (s & 7) * 4;
      float4 v = *(const float4*)(W + (size_t)(r0 + row) * CCH + k0 + k4);
      h16* d = As + row * ASTR + k4;
      d[0] = (h16)v.x; d[1] = (h16)v.y; d[2] = (h16)v.z; d[3] = (h16)v.w;
    }
#pragma unroll
    for (int it = 0; it < 4; ++it) {       // B: x 32x128
      int s = tid + it * 256;
      int kk = s >> 5, n4 = (s & 31) * 4;
      float4 v = *(const float4*)(xb + (size_t)(k0 + kk) * NPIX + n0 + n4);
      stB(Bs, kk, n4 + 0, (h16)v.x); stB(Bs, kk, n4 + 1, (h16)v.y);
      stB(Bs, kk, n4 + 2, (h16)v.z); stB(Bs, kk, n4 + 3, (h16)v.w);
    }
    __syncthreads();
    mma_tile(As, Bs, wm, wn, lane, acc);
    __syncthreads();
  }
  h16* po = (grp == 0) ? phi : (grp == 1) ? theta : g;
#pragma unroll
  for (int mt = 0; mt < 2; ++mt)
#pragma unroll
    for (int nt = 0; nt < 4; ++nt)
#pragma unroll
      for (int r = 0; r < 8; ++r) {
        int rg  = r0 + wm * 32 + mt * 16 + r + ((lane >> 4) << 3);
        int col = n0 + wn * 64 + nt * 16 + (lane & 15);
        float v = acc[mt][nt][r];
        size_t o = (size_t)(b * ICH + rg) * NPIX + col;
        if (grp < 3) po[o] = (h16)v;
        else         zx[o] = v;
      }
}

// ============================================================================
// Kernel 3: S = theta^T @ phi  (K=IC=256) -> f32 [B][N][N]
// grid: (N/128 cols m, N/128 rows n, B)
// ============================================================================
__global__ __launch_bounds__(256) void k_s(const h16* __restrict__ theta,
                                           const h16* __restrict__ phi,
                                           float* __restrict__ S) {
  __shared__ h16 As[TILE * ASTR];
  __shared__ h16 Bs[BSW];
  const int tid = threadIdx.x, lane = tid & 31, wid = tid >> 5;
  const int wm = wid & 3, wn = wid >> 2;
  const int b = blockIdx.z;
  const int n0 = blockIdx.y * TILE;   // output rows (theta index n)
  const int m0 = blockIdx.x * TILE;   // output cols (phi index m)
  const h16* tb = theta + (size_t)b * ICH * NPIX;
  const h16* pb = phi   + (size_t)b * ICH * NPIX;
  v8f acc[2][4] = {};
  for (int k0 = 0; k0 < ICH; k0 += KSTEP) {
#pragma unroll
    for (int it = 0; it < 16; ++it) {       // 4096 h16 elements each tile
      int s = tid + it * 256;
      int kk = s >> 7, nl = s & 127;        // coalesced over nl
      As[nl * ASTR + kk] = tb[(size_t)(k0 + kk) * NPIX + n0 + nl]; // transpose
      stB(Bs, kk, nl, pb[(size_t)(k0 + kk) * NPIX + m0 + nl]);
    }
    __syncthreads();
    mma_tile(As, Bs, wm, wn, lane, acc);
    __syncthreads();
  }
  float* Sb = S + (size_t)b * NPIX * NPIX;
#pragma unroll
  for (int mt = 0; mt < 2; ++mt)
#pragma unroll
    for (int nt = 0; nt < 4; ++nt)
#pragma unroll
      for (int r = 0; r < 8; ++r) {
        int row = n0 + wm * 32 + mt * 16 + r + ((lane >> 4) << 3);
        int col = m0 + wn * 64 + nt * 16 + (lane & 15);
        Sb[(size_t)row * NPIX + col] = acc[mt][nt][r];
      }
}

// ============================================================================
// Kernel 4: per-column (m) online softmax stats over n: cmax, csum
// ============================================================================
__global__ __launch_bounds__(256) void k_colstats(const float* __restrict__ S,
                                                  float* __restrict__ cmax,
                                                  float* __restrict__ csum) {
  const int b = blockIdx.y;
  const int m = blockIdx.x * 256 + threadIdx.x;
  const float* Sb = S + (size_t)b * NPIX * NPIX;
  float mx = -3.4e38f, sm = 0.f;
  for (int n = 0; n < NPIX; ++n) {
    __builtin_prefetch(Sb + (size_t)(n + 8) * NPIX + m, 0, 0);
    float v = Sb[(size_t)n * NPIX + m];
    if (v > mx) { sm *= __expf(mx - v); mx = v; }
    sm += __expf(v - mx);
  }
  cmax[b * NPIX + m] = mx;
  csum[b * NPIX + m] = sm;
}

// ============================================================================
// Kernel 5: gs[i][m] = g[i][m] / csum[m]   (folds softmax denominator into g)
// ============================================================================
__global__ void k_scaleg(const h16* __restrict__ g, const float* __restrict__ csum,
                         h16* __restrict__ gs) {
  size_t idx = (size_t)blockIdx.x * blockDim.x + threadIdx.x;
  if (idx >= (size_t)BB * ICH * NPIX) return;
  int m = (int)(idx & (NPIX - 1));
  int b = (int)(idx / ((size_t)ICH * NPIX));
  gs[idx] = (h16)((float)g[idx] / csum[b * NPIX + m]);
}

// ============================================================================
// Kernel 6: y[n][i] = sum_m exp(S[n][m]-cmax[m]) * gs[i][m]   (K=N=4096)
// grid: (IC/128, N/128, B); y stored f16 [B][N][IC]
// ============================================================================
__global__ __launch_bounds__(256) void k_y(const float* __restrict__ S,
                                           const float* __restrict__ cmax,
                                           const h16* __restrict__ gs,
                                           h16* __restrict__ y) {
  __shared__ h16 As[TILE * ASTR];
  __shared__ h16 Bs[BSW];
  __shared__ float cms[KSTEP];
  const int tid = threadIdx.x, lane = tid & 31, wid = tid >> 5;
  const int wm = wid & 3, wn = wid >> 2;
  const int b = blockIdx.z;
  const int n0 = blockIdx.y * TILE;   // rows n
  const int i0 = blockIdx.x * TILE;   // cols i
  const float* Sb  = S  + (size_t)b * NPIX * NPIX;
  const h16*   gb  = gs + (size_t)b * ICH * NPIX;
  const float* cmb = cmax + b * NPIX;
  v8f acc[2][4] = {};
  for (int m0 = 0; m0 < NPIX; m0 += KSTEP) {
    if (tid < KSTEP) cms[tid] = cmb[m0 + tid];
    __syncthreads();
#pragma unroll
    for (int it = 0; it < 4; ++it) {        // A: exp(S - cmax) 128x32, on the fly
      int s = tid + it * 256;
      int row = s >> 3, k4 = (s & 7) * 4;
      float4 v = *(const float4*)(Sb + (size_t)(n0 + row) * NPIX + m0 + k4);
      h16* d = As + row * ASTR + k4;
      d[0] = (h16)__expf(v.x - cms[k4 + 0]);
      d[1] = (h16)__expf(v.y - cms[k4 + 1]);
      d[2] = (h16)__expf(v.z - cms[k4 + 2]);
      d[3] = (h16)__expf(v.w - cms[k4 + 3]);
    }
#pragma unroll
    for (int it = 0; it < 16; ++it) {       // B: gs^T tile 32x128
      int s = tid + it * 256;
      int ii = s >> 5, mm = s & 31;
      stB(Bs, mm, ii, gb[(size_t)(i0 + ii) * NPIX + m0 + mm]);
    }
    __syncthreads();
    mma_tile(As, Bs, wm, wn, lane, acc);
    __syncthreads();
  }
#pragma unroll
  for (int mt = 0; mt < 2; ++mt)
#pragma unroll
    for (int nt = 0; nt < 4; ++nt)
#pragma unroll
      for (int r = 0; r < 8; ++r) {
        int row = n0 + wm * 32 + mt * 16 + r + ((lane >> 4) << 3);
        int col = i0 + wn * 64 + nt * 16 + (lane & 15);
        y[((size_t)b * NPIX + row) * ICH + col] = (h16)acc[mt][nt][r];
      }
}

// ============================================================================
// Kernel 7: z = Wcm @ y^T + zx ; BN(eval) + ReLU -> out f32 [B][256][N]
// grid: (N/128, 256/128, B)
// ============================================================================
__global__ __launch_bounds__(256) void k_final(const h16* __restrict__ Wcm,
    const h16* __restrict__ y, const float* __restrict__ zx,
    const float* __restrict__ gamma, const float* __restrict__ beta,
    const float* __restrict__ mean, const float* __restrict__ var,
    float* __restrict__ out) {
  __shared__ h16 As[TILE * ASTR];
  __shared__ h16 Bs[BSW];
  const int tid = threadIdx.x, lane = tid & 31, wid = tid >> 5;
  const int wm = wid & 3, wn = wid >> 2;
  const int b = blockIdx.z;
  const int o0 = blockIdx.y * TILE;
  const int n0 = blockIdx.x * TILE;
  const h16* yb = y + (size_t)b * NPIX * ICH;
  v8f acc[2][4] = {};
  for (int k0 = 0; k0 < ICH; k0 += KSTEP) {
#pragma unroll
    for (int it = 0; it < 16; ++it) {
      int s = tid + it * 256;
      int row = s >> 5, kk = s & 31;       // coalesced over kk
      As[row * ASTR + kk] = Wcm[(size_t)(o0 + row) * ICH + k0 + kk];
      stB(Bs, kk, row, yb[(size_t)(n0 + row) * ICH + k0 + kk]); // transpose
    }
    __syncthreads();
    mma_tile(As, Bs, wm, wn, lane, acc);
    __syncthreads();
  }
#pragma unroll
  for (int mt = 0; mt < 2; ++mt)
#pragma unroll
    for (int nt = 0; nt < 4; ++nt)
#pragma unroll
      for (int r = 0; r < 8; ++r) {
        int o = o0 + wm * 32 + mt * 16 + r + ((lane >> 4) << 3);
        int n = n0 + wn * 64 + nt * 16 + (lane & 15);
        size_t off = (size_t)(b * ICH + o) * NPIX + n;
        float z = acc[mt][nt][r] + zx[off];
        float inv = gamma[o] * rsqrtf(var[o] + 1e-5f);
        float zz = z * inv + (beta[o] - mean[o] * inv);
        out[off] = fmaxf(zz, 0.f);
      }
}

// ============================================================================
extern "C" void kernel_launch(void* const* d_in, const int* in_sizes, int n_in,
                              void* d_out, int out_size, void* d_ws, size_t ws_size,
                              hipStream_t stream) {
  (void)in_sizes; (void)n_in; (void)out_size; (void)ws_size;
  const float* x       = (const float*)d_in[0];
  const float* w_phi   = (const float*)d_in[1];
  const float* w_theta = (const float*)d_in[2];
  const float* w_g     = (const float*)d_in[3];
  const float* w_mask  = (const float*)d_in[4];
  const float* w_conv1 = (const float*)d_in[5];
  const float* bn_g    = (const float*)d_in[6];
  const float* bn_b    = (const float*)d_in[7];
  const float* bn_m    = (const float*)d_in[8];
  const float* bn_v    = (const float*)d_in[9];
  float* out = (float*)d_out;

  char* ws = (char*)d_ws;
  size_t off = 0;
  float* S    = (float*)(ws + off); off += (size_t)BB * NPIX * NPIX * 4;  // 128 MB
  h16*  phi   = (h16*)(ws + off);   off += (size_t)BB * ICH * NPIX * 2;   // 4 MB
  h16*  theta = (h16*)(ws + off);   off += (size_t)BB * ICH * NPIX * 2;
  h16*  g     = (h16*)(ws + off);   off += (size_t)BB * ICH * NPIX * 2;
  h16*  gsb   = (h16*)(ws + off);   off += (size_t)BB * ICH * NPIX * 2;
  float* zx   = (float*)(ws + off); off += (size_t)BB * ICH * NPIX * 4;   // 8 MB
  h16*  y     = (h16*)(ws + off);   off += (size_t)BB * NPIX * ICH * 2;
  float* cmax = (float*)(ws + off); off += (size_t)BB * NPIX * 4;
  float* csum = (float*)(ws + off); off += (size_t)BB * NPIX * 4;
  h16*  Wcm   = (h16*)(ws + off);   off += (size_t)ICH * ICH * 2;

  k_wcm    <<<dim3(2, 2, 1),   256, 0, stream>>>(w_conv1, w_mask, Wcm);
  k_proj   <<<dim3(32, 8, BB), 256, 0, stream>>>(x, w_phi, w_theta, w_g, w_conv1,
                                                 phi, theta, g, zx, 0);
  k_s      <<<dim3(32, 32, BB), 256, 0, stream>>>(theta, phi, S);
  k_colstats<<<dim3(16, BB, 1), 256, 0, stream>>>(S, cmax, csum);
  k_scaleg <<<dim3((BB * ICH * NPIX + 255) / 256, 1, 1), 256, 0, stream>>>(g, csum, gsb);
  k_y      <<<dim3(2, 32, BB), 256, 0, stream>>>(S, cmax, gsb, y);
  k_final  <<<dim3(32, 2, BB), 256, 0, stream>>>(Wcm, y, zx, bn_g, bn_b, bn_m, bn_v, out);
}
